// MoE_8246337208877
// MI455X (gfx1250) — matrix operand order repaired and checked by reading, not compiled, not run on hardware
//
#include <hip/hip_runtime.h>
#include <hip/hip_bf16.h>
#include <stdint.h>

#define TOK 2048   // B*L
#define DIM 2048   // D
#define NEXP 8     // E
#define HID 1408   // H
#define HSH 2816   // HS

typedef __attribute__((ext_vector_type(16))) __bf16 v16bf;
typedef __attribute__((ext_vector_type(8)))  __bf16 v8bf;
typedef __attribute__((ext_vector_type(8)))  float  v8f;

__device__ __forceinline__ unsigned int f2bf(float f) {
    union { float f; unsigned int u; } x; x.f = f;
    unsigned int u = x.u;
    unsigned int r = u + 0x7fffu + ((u >> 16) & 1u);  // round-to-nearest-even
    return r >> 16;
}

// pack two floats into two bf16 in one dword (a -> low half / lower K slot)
__device__ __forceinline__ unsigned int pack2bf(float a, float b) {
    return (f2bf(a) & 0xffffu) | (f2bf(b) << 16);
}

__device__ __forceinline__ v16bf joinbf(v8bf lo, v8bf hi) {
    union { v16bf v; v8bf h[2]; } u;
    u.h[0] = lo; u.h[1] = hi;
    return u.v;
}

__device__ __forceinline__ float fast_sigmoid(float z) {
    return __builtin_amdgcn_rcpf(1.f + __expf(-z));
}

// LDS layout index (bf16 element units) so each lane reads its 16x16x32 bf16
// B-fragment as one contiguous 32B vector: lane = (n%16)+16*((k/8)%2); slot = 8*(k/16)+k%8.
__device__ __forceinline__ int bidx(int k, int n) {
    return (((n >> 4) * 32) + (n & 15) + (((k >> 3) & 1) << 4)) * 16 + ((k >> 4) << 3) + (k & 7);
}

// ---------------------------------------------------------------- utilities
__global__ void k_zero_counts(int* counts) {
    if (threadIdx.x < NEXP) counts[threadIdx.x] = 0;
}

// packed f32 -> bf16 conversion: 8 elements / thread
__global__ void k_cvt_x(const float* __restrict__ x, unsigned int* __restrict__ xb32) {
    int i = blockIdx.x * 256 + threadIdx.x;        // grid covers TOK*DIM/8
    const float4 a = *(const float4*)&x[(size_t)i * 8];
    const float4 b = *(const float4*)&x[(size_t)i * 8 + 4];
    unsigned int* p = xb32 + (size_t)i * 4;
    p[0] = pack2bf(a.x, a.y);
    p[1] = pack2bf(a.z, a.w);
    p[2] = pack2bf(b.x, b.y);
    p[3] = pack2bf(b.z, b.w);
}

// ---------------------------------------------------------------- router
__global__ void k_router(const float* __restrict__ x, const float* __restrict__ gate_w,
                         const float* __restrict__ shgate_w,
                         int* __restrict__ counts, int* __restrict__ lists,
                         float* __restrict__ wslot, float* __restrict__ sgate) {
    int lane = threadIdx.x & 31;
    int wid  = threadIdx.x >> 5;
    int t = blockIdx.x * 8 + wid;
    if (t >= TOK) return;
    const float* xr = x + (size_t)t * DIM;
    float acc[NEXP + 1];
#pragma unroll
    for (int e = 0; e <= NEXP; ++e) acc[e] = 0.f;
    for (int d = lane; d < DIM; d += 32) {
        float xv = xr[d];
#pragma unroll
        for (int e = 0; e < NEXP; ++e) acc[e] += xv * gate_w[e * DIM + d];
        acc[NEXP] += xv * shgate_w[d];
    }
#pragma unroll
    for (int off = 16; off >= 1; off >>= 1) {
#pragma unroll
        for (int e = 0; e <= NEXP; ++e) acc[e] += __shfl_xor(acc[e], off, 32);
    }
    if (lane == 0) {
        int e0 = 0; float l0 = acc[0];
        for (int e = 1; e < NEXP; ++e) if (acc[e] > l0) { l0 = acc[e]; e0 = e; }
        int e1 = (e0 == 0) ? 1 : 0; float l1 = acc[e1];
        for (int e = 0; e < NEXP; ++e)
            if (e != e0 && acc[e] > l1) { l1 = acc[e]; e1 = e; }
        float m  = fmaxf(l0, l1);
        float p0 = __expf(l0 - m), p1 = __expf(l1 - m);
        float inv = __builtin_amdgcn_rcpf(p0 + p1);
        int pos0 = atomicAdd(&counts[e0], 1);
        lists[e0 * TOK + pos0] = t * 2;
        wslot[t * 2] = p0 * inv;
        int pos1 = atomicAdd(&counts[e1], 1);
        lists[e1 * TOK + pos1] = t * 2 + 1;
        wslot[t * 2 + 1] = p1 * inv;
        sgate[t] = fast_sigmoid(acc[NEXP]);
    }
}

// Stage one 32xK-step x 64-col f32 weight panel into LDS as bf16 in fragment
// order. Each thread: two float4 loads (rows k2,k2+1), four packed b32 stores.
#define STAGE_PANEL(W, NCOLS, LDS32, k0, k2, n4)                                  \
    {                                                                             \
        const float4 r0 = *(const float4*)&(W)[(size_t)((k0) + (k2)) * (NCOLS) + (n4)];     \
        const float4 r1 = *(const float4*)&(W)[(size_t)((k0) + (k2) + 1) * (NCOLS) + (n4)]; \
        (LDS32)[bidx((k2), (n4) + 0) >> 1] = pack2bf(r0.x, r1.x);                 \
        (LDS32)[bidx((k2), (n4) + 1) >> 1] = pack2bf(r0.y, r1.y);                 \
        (LDS32)[bidx((k2), (n4) + 2) >> 1] = pack2bf(r0.z, r1.z);                 \
        (LDS32)[bidx((k2), (n4) + 3) >> 1] = pack2bf(r0.w, r1.w);                 \
    }

// ---------------------------------------------------------------- expert gate/up
__global__ void k_expert_gu(const unsigned short* __restrict__ xb,
                            const float* __restrict__ w_gate,
                            const float* __restrict__ w_up,
                            const int* __restrict__ counts,
                            const int* __restrict__ lists,
                            unsigned short* __restrict__ hbuf) {
    __shared__ unsigned int ldsG[1024];
    __shared__ unsigned int ldsU[1024];
    int e   = blockIdx.z;
    int cnt = counts[e];
    int i0  = blockIdx.x * 32;
    if (i0 >= cnt) return;
    int n0  = blockIdx.y * 64;
    int tid = threadIdx.x, lane = tid & 31, wid = tid >> 5;
    int mi = wid & 1, ni = wid >> 1;
    int half = lane >> 4;
    int k2 = (tid >> 4) * 2;        // staging row pair 0..30
    int n4 = (tid & 15) * 4;        // staging col group

    int rowA = i0 + mi * 16 + (lane & 15);
    int tok = 0;
    if (rowA < cnt) tok = lists[e * TOK + rowA] >> 1;
    const unsigned short* aptr = xb + (size_t)tok * DIM;
    const float* wg = w_gate + (size_t)e * DIM * HID + n0;
    const float* wu = w_up   + (size_t)e * DIM * HID + n0;

    v8f cg = {}; v8f cu = {};
    for (int k0 = 0; k0 < DIM; k0 += 32) {
        __syncthreads();
        if (k0 + 32 < DIM) {
            __builtin_prefetch(&wg[(size_t)(k0 + 32 + k2) * HID + n4], 0, 3);
            __builtin_prefetch(&wu[(size_t)(k0 + 32 + k2) * HID + n4], 0, 3);
        }
        STAGE_PANEL(wg, HID, ldsG, k0, k2, n4);
        STAGE_PANEL(wu, HID, ldsU, k0, k2, n4);
        __syncthreads();
        v16bf a  = joinbf(*(const v8bf*)(aptr + k0 + half * 8),
                          *(const v8bf*)(aptr + k0 + 16 + half * 8));
        v16bf bg = *(const v16bf*)&ldsG[(ni * 32 + lane) * 8];
        v16bf bu = *(const v16bf*)&ldsU[(ni * 32 + lane) * 8];
        cg = __builtin_amdgcn_wmma_f32_16x16x32_bf16(false, a, false, bg, (short)0, cg, false, false);
        cu = __builtin_amdgcn_wmma_f32_16x16x32_bf16(false, a, false, bu, (short)0, cu, false, false);
    }
    int col = n0 + ni * 16 + (lane & 15);
#pragma unroll
    for (int r = 0; r < 8; ++r) {
        int m = r + (half << 3);
        int rowm = i0 + mi * 16 + m;
        if (rowm < cnt) {
            int slot = lists[e * TOK + rowm];
            float g = cg[r], u = cu[r];
            float h = g * fast_sigmoid(g) * u;   // silu(g)*u
            hbuf[(size_t)slot * HID + col] = (unsigned short)f2bf(h);
        }
    }
}

// ---------------------------------------------------------------- expert down
__global__ void k_expert_down(const unsigned short* __restrict__ hbuf,
                              const float* __restrict__ w_down,
                              const int* __restrict__ counts,
                              const int* __restrict__ lists,
                              float* __restrict__ ybuf) {
    __shared__ unsigned int ldsB[1024];
    int e   = blockIdx.z;
    int cnt = counts[e];
    int i0  = blockIdx.x * 32;
    if (i0 >= cnt) return;
    int n0  = blockIdx.y * 64;
    int tid = threadIdx.x, lane = tid & 31, wid = tid >> 5;
    int mi = wid & 1, ni = wid >> 1;
    int half = lane >> 4;
    int k2 = (tid >> 4) * 2;
    int n4 = (tid & 15) * 4;

    int rowA = i0 + mi * 16 + (lane & 15);
    int srow = 0;
    if (rowA < cnt) srow = lists[e * TOK + rowA];
    const unsigned short* aptr = hbuf + (size_t)srow * HID;
    const float* wd = w_down + (size_t)e * HID * DIM + n0;

    v8f c = {};
    for (int k0 = 0; k0 < HID; k0 += 32) {
        __syncthreads();
        if (k0 + 32 < HID)
            __builtin_prefetch(&wd[(size_t)(k0 + 32 + k2) * DIM + n4], 0, 3);
        STAGE_PANEL(wd, DIM, ldsB, k0, k2, n4);
        __syncthreads();
        v16bf a = joinbf(*(const v8bf*)(aptr + k0 + half * 8),
                         *(const v8bf*)(aptr + k0 + 16 + half * 8));
        v16bf b = *(const v16bf*)&ldsB[(ni * 32 + lane) * 8];
        c = __builtin_amdgcn_wmma_f32_16x16x32_bf16(false, a, false, b, (short)0, c, false, false);
    }
    int col = n0 + ni * 16 + (lane & 15);
#pragma unroll
    for (int r = 0; r < 8; ++r) {
        int m = r + (half << 3);
        int rowm = i0 + mi * 16 + m;
        if (rowm < cnt) {
            int slot = lists[e * TOK + rowm];
            ybuf[(size_t)slot * DIM + col] = c[r];
        }
    }
}

// ---------------------------------------------------------------- shared expert
__global__ void k_shared_gu(const unsigned short* __restrict__ xb,
                            const float* __restrict__ shw_gate,
                            const float* __restrict__ shw_up,
                            unsigned short* __restrict__ shh) {
    __shared__ unsigned int ldsG[1024];
    __shared__ unsigned int ldsU[1024];
    int t0 = blockIdx.x * 32;
    int n0 = blockIdx.y * 64;
    int tid = threadIdx.x, lane = tid & 31, wid = tid >> 5;
    int mi = wid & 1, ni = wid >> 1;
    int half = lane >> 4;
    int k2 = (tid >> 4) * 2;
    int n4 = (tid & 15) * 4;

    int rowA = t0 + mi * 16 + (lane & 15);
    const unsigned short* aptr = xb + (size_t)rowA * DIM;
    const float* wg = shw_gate + n0;
    const float* wu = shw_up   + n0;

    v8f cg = {}; v8f cu = {};
    for (int k0 = 0; k0 < DIM; k0 += 32) {
        __syncthreads();
        if (k0 + 32 < DIM) {
            __builtin_prefetch(&wg[(size_t)(k0 + 32 + k2) * HSH + n4], 0, 3);
            __builtin_prefetch(&wu[(size_t)(k0 + 32 + k2) * HSH + n4], 0, 3);
        }
        STAGE_PANEL(wg, HSH, ldsG, k0, k2, n4);
        STAGE_PANEL(wu, HSH, ldsU, k0, k2, n4);
        __syncthreads();
        v16bf a  = joinbf(*(const v8bf*)(aptr + k0 + half * 8),
                          *(const v8bf*)(aptr + k0 + 16 + half * 8));
        v16bf bg = *(const v16bf*)&ldsG[(ni * 32 + lane) * 8];
        v16bf bu = *(const v16bf*)&ldsU[(ni * 32 + lane) * 8];
        cg = __builtin_amdgcn_wmma_f32_16x16x32_bf16(false, a, false, bg, (short)0, cg, false, false);
        cu = __builtin_amdgcn_wmma_f32_16x16x32_bf16(false, a, false, bu, (short)0, cu, false, false);
    }
    int col = n0 + ni * 16 + (lane & 15);
#pragma unroll
    for (int r = 0; r < 8; ++r) {
        int m = r + (half << 3);
        int tm = t0 + mi * 16 + m;
        float g = cg[r], u = cu[r];
        float h = g * fast_sigmoid(g) * u;
        shh[(size_t)tm * HSH + col] = (unsigned short)f2bf(h);
    }
}

__global__ void k_shared_down(const unsigned short* __restrict__ shh,
                              const float* __restrict__ shw_down,
                              const float* __restrict__ sgate,
                              float* __restrict__ out) {
    __shared__ unsigned int ldsB[1024];
    int t0 = blockIdx.x * 32;
    int n0 = blockIdx.y * 64;
    int tid = threadIdx.x, lane = tid & 31, wid = tid >> 5;
    int mi = wid & 1, ni = wid >> 1;
    int half = lane >> 4;
    int k2 = (tid >> 4) * 2;
    int n4 = (tid & 15) * 4;

    int rowA = t0 + mi * 16 + (lane & 15);
    const unsigned short* aptr = shh + (size_t)rowA * HSH;
    const float* wd = shw_down + n0;

    v8f c = {};
    for (int k0 = 0; k0 < HSH; k0 += 32) {
        __syncthreads();
        if (k0 + 32 < HSH)
            __builtin_prefetch(&wd[(size_t)(k0 + 32 + k2) * DIM + n4], 0, 3);
        STAGE_PANEL(wd, DIM, ldsB, k0, k2, n4);
        __syncthreads();
        v16bf a = joinbf(*(const v8bf*)(aptr + k0 + half * 8),
                         *(const v8bf*)(aptr + k0 + 16 + half * 8));
        v16bf b = *(const v16bf*)&ldsB[(ni * 32 + lane) * 8];
        c = __builtin_amdgcn_wmma_f32_16x16x32_bf16(false, a, false, b, (short)0, c, false, false);
    }
    int col = n0 + ni * 16 + (lane & 15);
#pragma unroll
    for (int r = 0; r < 8; ++r) {
        int m = r + (half << 3);
        int tm = t0 + mi * 16 + m;
        out[(size_t)tm * DIM + col] = sgate[tm] * c[r];
    }
}

// ---------------------------------------------------------------- final combine
__global__ void k_combine(const float* __restrict__ ybuf,
                          const float* __restrict__ wslot,
                          float* __restrict__ out) {
    int i = blockIdx.x * 256 + threadIdx.x;
    if (i >= TOK * DIM) return;
    int t = i >> 11;        // DIM == 2048
    size_t s0 = (size_t)(t * 2) * DIM + (i & (DIM - 1));
    out[i] += wslot[t * 2] * ybuf[s0] + wslot[t * 2 + 1] * ybuf[s0 + DIM];
}

// ---------------------------------------------------------------- launch
extern "C" void kernel_launch(void* const* d_in, const int* in_sizes, int n_in,
                              void* d_out, int out_size, void* d_ws, size_t ws_size,
                              hipStream_t stream) {
    (void)in_sizes; (void)n_in; (void)out_size; (void)ws_size;

    const float* x        = (const float*)d_in[0];
    const float* gate_w   = (const float*)d_in[1];
    const float* w_gate   = (const float*)d_in[2];
    const float* w_up     = (const float*)d_in[3];
    const float* w_down   = (const float*)d_in[4];
    const float* shgate_w = (const float*)d_in[5];
    const float* shw_gate = (const float*)d_in[6];
    const float* shw_up   = (const float*)d_in[7];
    const float* shw_down = (const float*)d_in[8];
    float* out = (float*)d_out;

    char* ws = (char*)d_ws;
    size_t off = 0;
    auto take = [&](size_t bytes) -> void* {
        void* p = ws + off;
        off += (bytes + 255) & ~(size_t)255;
        return p;
    };
    int*            counts = (int*)take(NEXP * sizeof(int));
    int*            lists  = (int*)take((size_t)NEXP * TOK * sizeof(int));
    float*          wslot  = (float*)take((size_t)TOK * 2 * sizeof(float));
    float*          sgate  = (float*)take((size_t)TOK * sizeof(float));
    unsigned short* xb     = (unsigned short*)take((size_t)TOK * DIM * 2);
    unsigned short* hbuf   = (unsigned short*)take((size_t)TOK * 2 * HID * 2);
    unsigned short* shh    = (unsigned short*)take((size_t)TOK * HSH * 2);
    float*          ybuf   = (float*)take((size_t)TOK * 2 * DIM * sizeof(float));

    k_zero_counts<<<1, 32, 0, stream>>>(counts);
    k_cvt_x<<<(TOK * DIM) / (256 * 8), 256, 0, stream>>>(x, (unsigned int*)xb);
    k_router<<<TOK / 8, 256, 0, stream>>>(x, gate_w, shgate_w, counts, lists, wslot, sgate);
    k_expert_gu<<<dim3(TOK / 32, HID / 64, NEXP), 256, 0, stream>>>(xb, w_gate, w_up, counts, lists, hbuf);
    k_expert_down<<<dim3(TOK / 32, DIM / 64, NEXP), 256, 0, stream>>>(hbuf, w_down, counts, lists, ybuf);
    k_shared_gu<<<dim3(TOK / 32, HSH / 64), 256, 0, stream>>>(xb, shw_gate, shw_up, shh);
    k_shared_down<<<dim3(TOK / 32, DIM / 64), 256, 0, stream>>>(shh, shw_down, sgate, out);
    k_combine<<<(TOK * DIM) / 256, 256, 0, stream>>>(ybuf, wslot, out);
}
